// RNN_35682588295549
// MI455X (gfx1250) — matrix operand order repaired and checked by reading, not compiled, not run on hardware
//
#include <hip/hip_runtime.h>
#include <cstdint>
#include <cstddef>

// ---- problem constants (match reference) ------------------------------------
#define T_STEPS 2048
#define BATCH   64
#define EDIM    512
#define HDIM    512

// LDS row padding (in bf16 elements) to stagger banks while keeping 16B align.
#define HPAD 8

typedef __attribute__((ext_vector_type(16))) __bf16 v16bf;
typedef __attribute__((ext_vector_type(8)))  __bf16 bf16x8;
typedef __attribute__((ext_vector_type(8)))  float  v8f;

// -----------------------------------------------------------------------------
// Kernel A: one-shot fp32 -> bf16 conversion of both weight matrices into ws.
// -----------------------------------------------------------------------------
__global__ __launch_bounds__(256) void cvt_weights(const float* __restrict__ Wxh,
                                                   const float* __restrict__ Whh,
                                                   __bf16* __restrict__ WxhB,
                                                   __bf16* __restrict__ WhhB) {
  const int i = blockIdx.x * blockDim.x + threadIdx.x;
  if (i < EDIM * HDIM) {
    WxhB[i] = (__bf16)Wxh[i];
    WhhB[i] = (__bf16)Whh[i];
  }
}

// -----------------------------------------------------------------------------
// Kernel B: Xp = X @ W_xh + b_h   (M=131072, K=512, N=512), bf16 WMMA, f32 acc.
// Block tile: 64(M) x 128(N), 8 waves; each wave: one 16x16 A-row x 4 N-tiles.
// LDS rows padded (+8 bf16) so ds_load_b128 fragments are bank-conflict free.
// -----------------------------------------------------------------------------
__global__ __launch_bounds__(256) void xproj(const float* __restrict__ X,
                                             const __bf16* __restrict__ Wb,
                                             const float* __restrict__ bh,
                                             float* __restrict__ Xp) {
  __shared__ __bf16 As[64][32 + HPAD];    // M x K tile, 80B rows (20-bank stagger)
  __shared__ __bf16 Bs[32][128 + HPAD];   // K x N tile, 272B rows (4-bank stagger)

  const int tid   = threadIdx.x;
  const int lane  = tid & 31;
  const int wid   = tid >> 5;
  const int l15   = lane & 15;
  const int lhalf = lane >> 4;
  const int m0    = blockIdx.x * 64;
  const int n0    = blockIdx.y * 128;
  const int wm    = (wid >> 1) * 16;   // 0,16,32,48
  const int wn    = (wid & 1)  * 64;   // 0,64

  v8f c[4];
#pragma unroll
  for (int j = 0; j < 4; ++j) {
    const float bv = bh[n0 + wn + j * 16 + l15];
#pragma unroll
    for (int v = 0; v < 8; ++v) c[j][v] = bv;
  }

  for (int k0 = 0; k0 < EDIM; k0 += 32) {
    // cooperative A load: 64x32 fp32 -> bf16 LDS (512 float4 chunks)
#pragma unroll
    for (int it = 0; it < 2; ++it) {
      const int chunk = tid + it * 256;
      const int r  = chunk >> 3;
      const int c4 = (chunk & 7) * 4;
      const float4 v = *(const float4*)(X + (size_t)(m0 + r) * EDIM + k0 + c4);
      As[r][c4 + 0] = (__bf16)v.x; As[r][c4 + 1] = (__bf16)v.y;
      As[r][c4 + 2] = (__bf16)v.z; As[r][c4 + 3] = (__bf16)v.w;
    }
    // cooperative B load: 32x128 bf16 (512 chunks of 8 bf16)
#pragma unroll
    for (int it = 0; it < 2; ++it) {
      const int chunk = tid + it * 256;
      const int r  = chunk >> 4;
      const int c8 = (chunk & 15) * 8;
      *(bf16x8*)&Bs[r][c8] = *(const bf16x8*)(Wb + (size_t)(k0 + r) * HDIM + n0 + c8);
    }
    __syncthreads();

    // A fragment: lane holds row M=l15, K in {0..7,16..23} (lhalf=0) / {8..15,24..31}
    v16bf a;
    {
      const bf16x8 alo = *(const bf16x8*)&As[wm + l15][lhalf * 8];
      const bf16x8 ahi = *(const bf16x8*)&As[wm + l15][16 + lhalf * 8];
#pragma unroll
      for (int i = 0; i < 8; ++i) { a[i] = alo[i]; a[i + 8] = ahi[i]; }
    }
#pragma unroll
    for (int j = 0; j < 4; ++j) {
      // B fragment: lane holds column N=l15, K in {0..7,16..23} (lhalf=0) / {8..15,24..31}
      v16bf b;
#pragma unroll
      for (int i = 0; i < 8; ++i) { b[i] = Bs[lhalf * 8 + i][wn + j * 16 + l15]; b[i + 8] = Bs[16 + lhalf * 8 + i][wn + j * 16 + l15]; }
      c[j] = __builtin_amdgcn_wmma_f32_16x16x32_bf16(false, a, false, b,
                                                     (short)0, c[j], false, false);
    }
    __syncthreads();
  }

  // C/D layout: lanes 0-15 -> N=lane, M=v; lanes 16-31 -> N=lane-16, M=v+8
#pragma unroll
  for (int j = 0; j < 4; ++j)
#pragma unroll
    for (int v = 0; v < 8; ++v) {
      const int m = v + 8 * lhalf;
      Xp[(size_t)(m0 + wm + m) * HDIM + n0 + wn + j * 16 + l15] = c[j][v];
    }
}

// -----------------------------------------------------------------------------
// Kernel C: sequential scan  h = tanh(Xp[t] + h @ W_hh).
// Batch split: 4 WGs x 16 batch rows -> NO cross-WG sync across the 2048 steps.
// h-slice lives in LDS (double-buffered, bank-staggered bf16); W_hh streams
// from L2 every step. FUSED=true computes X[t] @ W_xh + b_h in-loop (small ws).
// -----------------------------------------------------------------------------
template <bool FUSED>
__global__ __launch_bounds__(256) void rnn_scan(const float* __restrict__ X,
                                                const float* __restrict__ Xp,
                                                const __bf16* __restrict__ WxhB,
                                                const __bf16* __restrict__ WhhB,
                                                const float* __restrict__ bh,
                                                float* __restrict__ out,
                                                float* __restrict__ hlast) {
  __shared__ __bf16 hbuf[2][16][HDIM + HPAD];   // double-buffered hidden slice
  __shared__ __bf16 xbuf[16][EDIM + HPAD];      // only used when FUSED

  const int tid   = threadIdx.x;
  const int lane  = tid & 31;
  const int wid   = tid >> 5;
  const int l15   = lane & 15;
  const int lhalf = lane >> 4;
  const int r0    = blockIdx.x * 16;     // batch-row slice
  const int nb0   = wid * 64;            // this wave's 64 output columns

  for (int i = tid; i < 2 * 16 * (HDIM + HPAD); i += 256)
    (&hbuf[0][0][0])[i] = (__bf16)0.0f;
  __syncthreads();

  int cur = 0;
  for (int t = 0; t < T_STEPS; ++t) {
    v8f c[4];

    if (FUSED) {
      // stage X[t, r0:r0+16, :] as bf16 in LDS
      for (int i = tid; i < 16 * EDIM / 4; i += 256) {
        const int r  = i >> 7;
        const int c4 = (i & 127) * 4;
        const float4 v = *(const float4*)(X + ((size_t)t * BATCH + r0 + r) * EDIM + c4);
        xbuf[r][c4 + 0] = (__bf16)v.x; xbuf[r][c4 + 1] = (__bf16)v.y;
        xbuf[r][c4 + 2] = (__bf16)v.z; xbuf[r][c4 + 3] = (__bf16)v.w;
      }
      __syncthreads();
      // prefetch next step's X tile while we compute
      if (t + 1 < T_STEPS)
        __builtin_prefetch(X + ((size_t)(t + 1) * BATCH + r0 + l15) * EDIM + lhalf * 256, 0, 1);
#pragma unroll
      for (int j = 0; j < 4; ++j) {
        const float bv = bh[nb0 + j * 16 + l15];
#pragma unroll
        for (int v = 0; v < 8; ++v) c[j][v] = bv;
      }
#pragma unroll 4
      for (int k0 = 0; k0 < EDIM; k0 += 32) {
        // branch-free prefetch of the next K chunk (wraps on the last chunk)
        __builtin_prefetch(WxhB + (size_t)(((k0 + 32) & (EDIM - 1)) + lane) * HDIM + nb0, 0, 1);
        v16bf a;
        const bf16x8 alo = *(const bf16x8*)&xbuf[l15][k0 + lhalf * 8];
        const bf16x8 ahi = *(const bf16x8*)&xbuf[l15][k0 + 16 + lhalf * 8];
#pragma unroll
        for (int i = 0; i < 8; ++i) { a[i] = alo[i]; a[i + 8] = ahi[i]; }
#pragma unroll
        for (int j = 0; j < 4; ++j) {
          const __bf16* bp = WxhB + (size_t)k0 * HDIM + nb0 + j * 16 + l15;
          v16bf b;
#pragma unroll
          for (int i = 0; i < 8; ++i) { b[i] = bp[(size_t)(lhalf * 8 + i) * HDIM]; b[i + 8] = bp[(size_t)(16 + lhalf * 8 + i) * HDIM]; }
          c[j] = __builtin_amdgcn_wmma_f32_16x16x32_bf16(false, a, false, b,
                                                         (short)0, c[j], false, false);
        }
      }
    } else {
      // accumulator init = precomputed input projection (already includes bias)
      const float* xp = Xp + ((size_t)t * BATCH + r0) * HDIM;
#pragma unroll
      for (int j = 0; j < 4; ++j)
#pragma unroll
        for (int v = 0; v < 8; ++v)
          c[j][v] = xp[(size_t)(v + 8 * lhalf) * HDIM + nb0 + j * 16 + l15];
      // prefetch next step's Xp tile while we compute
      if (t + 1 < T_STEPS)
        __builtin_prefetch(xp + (size_t)BATCH * HDIM + (size_t)l15 * HDIM + nb0 + lhalf * 32, 0, 1);
    }

    // h @ W_hh, K = 512 in chunks of 32
#pragma unroll 4
    for (int k0 = 0; k0 < HDIM; k0 += 32) {
      // branch-free prefetch of the next K chunk (wraps on the last chunk)
      __builtin_prefetch(WhhB + (size_t)(((k0 + 32) & (HDIM - 1)) + lane) * HDIM + nb0, 0, 1);
      v16bf a;
      const bf16x8 alo = *(const bf16x8*)&hbuf[cur][l15][k0 + lhalf * 8];
      const bf16x8 ahi = *(const bf16x8*)&hbuf[cur][l15][k0 + 16 + lhalf * 8];
#pragma unroll
      for (int i = 0; i < 8; ++i) { a[i] = alo[i]; a[i + 8] = ahi[i]; }
#pragma unroll
      for (int j = 0; j < 4; ++j) {
        const __bf16* bp = WhhB + (size_t)k0 * HDIM + nb0 + j * 16 + l15;
        v16bf b;
#pragma unroll
        for (int i = 0; i < 8; ++i) { b[i] = bp[(size_t)(lhalf * 8 + i) * HDIM]; b[i + 8] = bp[(size_t)(16 + lhalf * 8 + i) * HDIM]; }
        c[j] = __builtin_amdgcn_wmma_f32_16x16x32_bf16(false, a, false, b,
                                                       (short)0, c[j], false, false);
      }
    }

    __syncthreads();  // all reads of hbuf[cur] complete before writing hbuf[nxt]

    float* dst = out + (size_t)t * BATCH * HDIM;
    const int nxt = cur ^ 1;
#pragma unroll
    for (int j = 0; j < 4; ++j)
#pragma unroll
      for (int v = 0; v < 8; ++v) {
        const int m = v + 8 * lhalf;
        const int n = nb0 + j * 16 + l15;
        const float val = tanhf(c[j][v]);
        dst[(size_t)(r0 + m) * HDIM + n] = val;
        hbuf[nxt][m][n] = (__bf16)val;
        if (t == T_STEPS - 1) hlast[(size_t)(r0 + m) * HDIM + n] = val;
      }
    __syncthreads();  // hbuf[nxt] writes visible before next step's reads
    cur = nxt;
  }
}

// -----------------------------------------------------------------------------
extern "C" void kernel_launch(void* const* d_in, const int* in_sizes, int n_in,
                              void* d_out, int out_size, void* d_ws, size_t ws_size,
                              hipStream_t stream) {
  const float* X   = (const float*)d_in[0];   // [T, B, E]
  const float* Wxh = (const float*)d_in[1];   // [E, H]
  const float* Whh = (const float*)d_in[2];   // [H, H]
  const float* bh  = (const float*)d_in[3];   // [H]

  float* out   = (float*)d_out;                              // hidden_states [T,B,H]
  float* hlast = out + (size_t)T_STEPS * BATCH * HDIM;       // h_last [B,H]

  // workspace layout: [W_xh bf16 | W_hh bf16 | Xp fp32]
  __bf16* WxhB = (__bf16*)d_ws;
  __bf16* WhhB = WxhB + (size_t)EDIM * HDIM;
  float*  Xp   = (float*)((char*)d_ws + (size_t)2 * EDIM * HDIM * sizeof(__bf16));
  const size_t need = (size_t)2 * EDIM * HDIM * sizeof(__bf16)
                    + (size_t)T_STEPS * BATCH * HDIM * sizeof(float);

  cvt_weights<<<dim3((EDIM * HDIM + 255) / 256), dim3(256), 0, stream>>>(
      Wxh, Whh, WxhB, WhhB);

  if (ws_size >= need) {
    // fast path: one big parallel input-projection GEMM, then the scan
    xproj<<<dim3((T_STEPS * BATCH) / 64, HDIM / 128), dim3(256), 0, stream>>>(
        X, WxhB, bh, Xp);
    rnn_scan<false><<<dim3(BATCH / 16), dim3(256), 0, stream>>>(
        X, Xp, WxhB, WhhB, bh, out, hlast);
  } else {
    // small-workspace path: fuse the input projection into the scan
    rnn_scan<true><<<dim3(BATCH / 16), dim3(256), 0, stream>>>(
        X, nullptr, WxhB, WhhB, bh, out, hlast);
  }
}
